// LSTMX1Series_X2Indices_21148418966324
// MI455X (gfx1250) — matrix operand order repaired and checked by reading, not compiled, not run on hardware
//
#include <hip/hip_runtime.h>
#include <math.h>

#define DEVINL __device__ __forceinline__

typedef __attribute__((ext_vector_type(16))) __bf16 bf16x16;
typedef __attribute__((ext_vector_type(8)))  __bf16 bf16x8;
typedef __attribute__((ext_vector_type(8)))  float  f32x8;

union ABFrag { bf16x16 v; bf16x8 h[2]; };

static constexpr int BATCH = 64;
static constexpr int SEQ   = 512;
static constexpr int HID   = 512;
static constexpr int BH    = BATCH * HID;
static constexpr int NWG   = 128;          // H / HC
static constexpr int HC    = 4;            // hidden units per workgroup
static constexpr int GR    = 4 * HC;       // gate rows per workgroup (i,f,g,o)

// ---------------------------------------------------------------- activations
DEVINL float sigmoidf_(float x) { return 1.0f / (1.0f + __expf(-x)); }
DEVINL float tanhf_(float x) {
  // 1 - 2/(e^{2x}+1): saturates correctly at +-1 without inf/inf
  float e = __expf(2.0f * x);
  return 1.0f - 2.0f / (e + 1.0f);
}
DEVINL unsigned pack2bf(float a, float b) {
  union { unsigned u; __bf16 h[2]; } r;
  r.h[0] = (__bf16)a; r.h[1] = (__bf16)b;
  return r.u;
}

// ---------------------------------------------------------------- grid barrier
DEVINL void grid_barrier(unsigned* cnt, unsigned target) {
  __syncthreads();
  if (threadIdx.x == 0) {
    __hip_atomic_fetch_add(cnt, 1u, __ATOMIC_RELEASE, __HIP_MEMORY_SCOPE_AGENT);
    unsigned v;
    do {
      __builtin_amdgcn_s_sleep(1);
      v = __hip_atomic_load(cnt, __ATOMIC_ACQUIRE, __HIP_MEMORY_SCOPE_AGENT);
    } while (v < target);
  }
  __syncthreads();
}

// ---------------------------------------------------------------- prep kernels
__global__ void k_xcat(const float* __restrict__ prim, const float* __restrict__ aux,
                       __bf16* __restrict__ xcat, int total) {
  for (int i = blockIdx.x * blockDim.x + threadIdx.x; i < total; i += gridDim.x * blockDim.x) {
    int k  = i % 96;
    int bt = i / 96;           // b*SEQ + t
    int b  = bt / SEQ;
    float v = (k < 64) ? prim[(size_t)bt * 64 + k] : aux[b * 32 + (k - 64)];
    xcat[i] = (__bf16)v;
  }
}

__global__ void k_wcat(const float* __restrict__ wih, const float* __restrict__ whh,
                       __bf16* __restrict__ wc, int K1, int total) {
  int K = K1 + HID;
  for (int i = blockIdx.x * blockDim.x + threadIdx.x; i < total; i += gridDim.x * blockDim.x) {
    int r = i / K, k = i % K;
    float v = (k < K1) ? wih[(size_t)r * K1 + k] : whh[(size_t)r * HID + (k - K1)];
    wc[i] = (__bf16)v;
  }
}

__global__ void k_bias(const float* __restrict__ a, const float* __restrict__ b,
                       float* __restrict__ o, int n) {
  for (int i = blockIdx.x * blockDim.x + threadIdx.x; i < n; i += gridDim.x * blockDim.x)
    o[i] = a[i] + b[i];
}

__global__ void k_cvt(const float* __restrict__ s, __bf16* __restrict__ d, int n) {
  for (int i = blockIdx.x * blockDim.x + threadIdx.x; i < n; i += gridDim.x * blockDim.x)
    d[i] = (__bf16)s[i];
}

__global__ void k_zero16(__bf16* __restrict__ d, int n) {
  for (int i = blockIdx.x * blockDim.x + threadIdx.x; i < n; i += gridDim.x * blockDim.x)
    d[i] = (__bf16)0.0f;
}

__global__ void k_zero32(unsigned* __restrict__ d, int n) {
  for (int i = blockIdx.x * blockDim.x + threadIdx.x; i < n; i += gridDim.x * blockDim.x)
    d[i] = 0u;
}

// ---------------------------------------------------------------- recurrence
// gates = [x_t | h_{t-1}] @ Wcat^T + bias ; c,h update ; persistent over 512 steps.
// grid = NWG(128) blocks, block = 128 threads (4 waves, wave32).
// Each WG owns HC=4 hidden units -> GR=16 gate rows, M=64 batch, N=16: 4 M-tiles, one per wave.
template <int K1>
__global__ __launch_bounds__(128) void lstm_recur(
    const __bf16* __restrict__ xsrc,   // [(b*SEQ+t)*K1 + k]
    __bf16* __restrict__ hbuf,         // double buffer 2*BH
    const __bf16* __restrict__ wcat,   // 2048 x (K1+512), row-major
    const float* __restrict__ bias,    // 2048 (b_ih + b_hh)
    __bf16* __restrict__ hseq,         // B*SEQ*HID output h sequence (may be null)
    unsigned* __restrict__ cnt) {
  constexpr int K = K1 + HID;
  extern __shared__ char smem[];
  __bf16* Wl    = (__bf16*)smem;                     // GR x K
  float*  biasl = (float*)(smem + (size_t)GR * K * 2);
  float*  gbuf  = biasl + GR;                        // 64 x GR

  const int tid = threadIdx.x;
  const int wg  = blockIdx.x;

  // ---- stage weight slice (rows: i,f,g,o blocks for this WG's hidden units)
  constexpr int CH = K / 8;
  for (int idx = tid; idx < GR * CH; idx += 128) {
    int r = idx / CH, c = idx % CH;
    int q = r >> 2, hc = wg * HC + (r & 3);
    const bf16x8* src = (const bf16x8*)(wcat + ((size_t)(q * HID + hc)) * K + c * 8);
    *(bf16x8*)(Wl + (size_t)r * K + c * 8) = *src;
  }
  if (tid < GR) {
    int q = tid >> 2, hc = wg * HC + (tid & 3);
    biasl[tid] = bias[q * HID + hc];
  }
  __syncthreads();

  const int lane  = tid & 31;
  const int wv    = tid >> 5;              // 0..3 -> M tile
  const int mbase = wv * 16;
  const int arow  = mbase + (lane & 15);   // batch row for A fragment
  const int aoff  = (lane >> 4) * 8;       // A k sub-offset
  const int boff  = (lane >> 4) * 8;       // B k sub-offset
  const int brow  = lane & 15;             // local W row for B fragment
  const int crow0 = mbase + (lane >> 4) * 8;
  const int gcol  = lane & 15;

  // elementwise mapping: 2 cell elements per thread
  const int e0   = tid * 2;
  const int b_e  = e0 >> 2;                // batch index
  const int hcl  = e0 & 3;                 // 0 or 2 (pairs share batch row)
  const int hidx = wg * HC + hcl;
  float c0 = 0.0f, c1 = 0.0f;

  for (int t = 0; t < SEQ; ++t) {
    const __bf16* hread  = hbuf + (size_t)(t & 1) * BH;
    __bf16*       hwrite = hbuf + (size_t)((t + 1) & 1) * BH;
    const __bf16* xrow   = xsrc + ((size_t)arow * SEQ + t) * K1;

    f32x8 acc = {0.f, 0.f, 0.f, 0.f, 0.f, 0.f, 0.f, 0.f};
    for (int kt = 0; kt < K / 32; ++kt) {
      const int k0 = kt * 32;
      ABFrag a, b;
      const __bf16* ap = (k0 < K1) ? (xrow + k0 + aoff)
                                   : (hread + (size_t)arow * HID + (k0 - K1) + aoff);
      a.h[0] = *(const bf16x8*)(ap);
      a.h[1] = *(const bf16x8*)(ap + 16);
      const __bf16* bp = Wl + (size_t)brow * K + k0 + boff;
      b.h[0] = *(const bf16x8*)(bp);
      b.h[1] = *(const bf16x8*)(bp + 16);
      acc = __builtin_amdgcn_wmma_f32_16x16x32_bf16(false, a.v, false, b.v,
                                                    (short)0, acc, false, false);
    }

    // gate pre-activations -> LDS
#pragma unroll
    for (int j = 0; j < 8; ++j)
      gbuf[(crow0 + j) * GR + gcol] = acc[j] + biasl[gcol];
    __syncthreads();

    // cell update (2 elements per thread)
    float hv[2];
#pragma unroll
    for (int u = 0; u < 2; ++u) {
      const float* g = gbuf + (size_t)b_e * GR + (hcl + u);
      float gi = sigmoidf_(g[0]);
      float gf = sigmoidf_(g[HC]);
      float gg = tanhf_(g[2 * HC]);
      float go = sigmoidf_(g[3 * HC]);
      float& c = u ? c1 : c0;
      c = gf * c + gi * gg;
      hv[u] = go * tanhf_(c);
    }
    unsigned hp = pack2bf(hv[0], hv[1]);
    *(unsigned*)(hwrite + (size_t)b_e * HID + hidx) = hp;
    if (hseq) *(unsigned*)(hseq + ((size_t)b_e * SEQ + t) * HID + hidx) = hp;

    grid_barrier(cnt, (unsigned)NWG * (unsigned)(t + 1));
  }
}

// ---------------------------------------------------------------- output GEMM
// out(64x128) = h(64x512 bf16) @ Wout^T + b_out ; one block, 8 waves, wave w -> N tile w.
__global__ __launch_bounds__(256) void out_gemm(const __bf16* __restrict__ h,
                                                const __bf16* __restrict__ wb,
                                                const float* __restrict__ bout,
                                                float* __restrict__ out) {
  const int tid = threadIdx.x, lane = tid & 31, wv = tid >> 5;
  const int nbase = wv * 16;
  const int aoff = (lane >> 4) * 8, boff = (lane >> 4) * 8;
  const int nrow = nbase + (lane & 15);
  const int ncol = nbase + (lane & 15);
  for (int mt = 0; mt < 4; ++mt) {
    const int arow = mt * 16 + (lane & 15);
    f32x8 acc = {0.f, 0.f, 0.f, 0.f, 0.f, 0.f, 0.f, 0.f};
    for (int kt = 0; kt < HID / 32; ++kt) {
      const int k0 = kt * 32;
      ABFrag a, b;
      const __bf16* ap = h + (size_t)arow * HID + k0 + aoff;
      a.h[0] = *(const bf16x8*)(ap);
      a.h[1] = *(const bf16x8*)(ap + 16);
      const __bf16* bp = wb + (size_t)nrow * HID + k0 + boff;
      b.h[0] = *(const bf16x8*)(bp);
      b.h[1] = *(const bf16x8*)(bp + 16);
      acc = __builtin_amdgcn_wmma_f32_16x16x32_bf16(false, a.v, false, b.v,
                                                    (short)0, acc, false, false);
    }
#pragma unroll
    for (int j = 0; j < 8; ++j) {
      int m = mt * 16 + (lane >> 4) * 8 + j;
      out[m * 128 + ncol] = acc[j] + bout[ncol];
    }
  }
}

// ---------------------------------------------------------------- launch
extern "C" void kernel_launch(void* const* d_in, const int* in_sizes, int n_in,
                              void* d_out, int out_size, void* d_ws, size_t ws_size,
                              hipStream_t stream) {
  (void)in_sizes; (void)n_in; (void)out_size; (void)ws_size;
  const float* prim = (const float*)d_in[0];
  const float* aux  = (const float*)d_in[1];
  const float* Wih0 = (const float*)d_in[2];
  const float* Whh0 = (const float*)d_in[3];
  const float* bih0 = (const float*)d_in[4];
  const float* bhh0 = (const float*)d_in[5];
  const float* Wih1 = (const float*)d_in[6];
  const float* Whh1 = (const float*)d_in[7];
  const float* bih1 = (const float*)d_in[8];
  const float* bhh1 = (const float*)d_in[9];
  const float* Wout = (const float*)d_in[10];
  const float* bout = (const float*)d_in[11];
  float* out = (float*)d_out;

  char* ws = (char*)d_ws;
  size_t o = 0;
  auto take = [&](size_t bytes) -> char* {
    char* p = ws + o;
    o = (o + bytes + 255) & ~(size_t)255;
    return p;
  };
  unsigned* cnt  = (unsigned*)take(256);                        // [0]=L0, [1]=L1
  __bf16* xcat   = (__bf16*)take((size_t)BATCH * SEQ * 96 * 2); // [x_t|aux] bf16
  __bf16* wcat0  = (__bf16*)take((size_t)2048 * 608 * 2);       // [Wih0|Whh0]
  __bf16* wcat1  = (__bf16*)take((size_t)2048 * 1024 * 2);      // [Wih1|Whh1]
  float*  bias0  = (float*)take(2048 * 4);
  float*  bias1  = (float*)take(2048 * 4);
  __bf16* hb0    = (__bf16*)take((size_t)2 * BH * 2);           // double-buffered h, layer0
  __bf16* hb1    = (__bf16*)take((size_t)2 * BH * 2);           // double-buffered h, layer1
  __bf16* h0seq  = (__bf16*)take((size_t)BATCH * SEQ * HID * 2);// layer-0 h sequence
  __bf16* woutb  = (__bf16*)take((size_t)128 * HID * 2);

  int tot;
  tot = BATCH * SEQ * 96;
  k_xcat<<<2048, 256, 0, stream>>>(prim, aux, xcat, tot);
  tot = 2048 * 608;
  k_wcat<<<2048, 256, 0, stream>>>(Wih0, Whh0, wcat0, 96, tot);
  tot = 2048 * 1024;
  k_wcat<<<2048, 256, 0, stream>>>(Wih1, Whh1, wcat1, 512, tot);
  k_bias<<<8, 256, 0, stream>>>(bih0, bhh0, bias0, 2048);
  k_bias<<<8, 256, 0, stream>>>(bih1, bhh1, bias1, 2048);
  k_cvt<<<256, 256, 0, stream>>>(Wout, woutb, 128 * HID);
  k_zero16<<<64, 256, 0, stream>>>(hb0, 2 * BH);
  k_zero16<<<64, 256, 0, stream>>>(hb1, 2 * BH);
  k_zero32<<<1, 64, 0, stream>>>(cnt, 2);

  const size_t smem0 = (size_t)GR * (96 + HID) * 2 + GR * 4 + (size_t)64 * GR * 4;
  lstm_recur<96><<<NWG, 128, smem0, stream>>>(xcat, hb0, wcat0, bias0, h0seq, cnt + 0);

  const size_t smem1 = (size_t)GR * (512 + HID) * 2 + GR * 4 + (size_t)64 * GR * 4;
  lstm_recur<512><<<NWG, 128, smem1, stream>>>(h0seq, hb1, wcat1, bias1, nullptr, cnt + 1);

  // 512 steps even -> final h lives in buffer 0
  out_gemm<<<1, 256, 0, stream>>>(hb1, woutb, bout, out);
}
